// OrderCourierHeteroGNN_23373212025402
// MI455X (gfx1250) — matrix operand order repaired and checked by reading, not compiled, not run on hardware
//
#include <hip/hip_runtime.h>
#include <math.h>

typedef __attribute__((ext_vector_type(16))) _Float16 v16h;
typedef __attribute__((ext_vector_type(8)))  float    v8f;
typedef __attribute__((ext_vector_type(4)))  float    v4f;

#define NOQ   100000
#define NRQ   1000
#define EQ    1000000
#define DOQ   64
#define DRQ   32
#define DEQ   4
#define HQ    64
#define OMQ   32
#define SCALEQ 0.125f   // 1/sqrt(64)

// ---------------- helpers ----------------
__device__ __forceinline__ unsigned f2ord(float f) {
    unsigned u = __float_as_uint(f);
    return (u & 0x80000000u) ? ~u : (u | 0x80000000u);
}
__device__ __forceinline__ float ord2f(unsigned u) {
    return (u & 0x80000000u) ? __uint_as_float(u & 0x7FFFFFFFu) : __uint_as_float(~u);
}

// ---------------- K0: weight conversion + swizzle into WMMA B-fragment layout --------
// Fragment layout (f16, K=32 chunk, 16x16 N tile):
//   lane L<16  : col = nt*16+L,  halves h -> K = kc*32 + (h>>3)*16 + (h&7)
//   lane L>=16 : col = nt*16+L-16, halves h -> K = kc*32 + (h>>3)*16 + 8 + (h&7)
// Stored so each lane's 16 halves are contiguous: idx = ((nt*KC + kc)*32 + lane)*16 + h
__global__ void convert_weights(const float* __restrict__ Wk, const float* __restrict__ Wv,
                                const float* __restrict__ Wp, const float* __restrict__ W1,
                                _Float16* Wkf, _Float16* Wvf, _Float16* Wpf, _Float16* W1f) {
    int i = blockIdx.x * blockDim.x + threadIdx.x;
    int n = blockDim.x * gridDim.x;
    // 64x64 matrices: nt in [0,4), kc in [0,2)  -> 8 fragments * 512 halves = 4096
    for (int t = i; t < 4096; t += n) {
        int h    = t & 15;
        int lane = (t >> 4) & 31;
        int f    = t >> 9;          // nt*2 + kc
        int kc   = f & 1;
        int nt   = f >> 1;
        int K    = kc * 32 + (h >> 3) * 16 + (lane >> 4) * 8 + (h & 7);
        int N    = nt * 16 + (lane & 15);
        Wkf[t] = (_Float16)Wk[K * 64 + N];
        Wvf[t] = (_Float16)Wv[K * 64 + N];
        Wpf[t] = (_Float16)Wp[K * 64 + N];
    }
    // W1 [160,128]: nt in [0,8), kc in [0,5) -> 40 fragments * 512 halves = 20480
    for (int t = i; t < 20480; t += n) {
        int h    = t & 15;
        int lane = (t >> 4) & 31;
        int f    = t >> 9;          // nt*5 + kc
        int kc   = f % 5;
        int nt   = f / 5;
        int K    = kc * 32 + (h >> 3) * 16 + (lane >> 4) * 8 + (h & 7);
        int N    = nt * 16 + (lane & 15);
        W1f[t] = (_Float16)W1[K * 128 + N];
    }
}

// ---------------- K1: init accumulators ----------------
__global__ void init_ws(unsigned* rmax, float* denom, float* aggnum) {
    int i = blockIdx.x * blockDim.x + threadIdx.x;
    if (i < NRQ) { rmax[i] = f2ord(-INFINITY); denom[i] = 0.0f; }
    if (i < NRQ * HQ) aggnum[i] = 0.0f;
}

// ---------------- K2: rider-side projections (tiny) ----------------
__global__ void rider_prep(const float* __restrict__ xr,
                           const float* __restrict__ Wq, const float* __restrict__ bq,
                           const float* __restrict__ Wskip, const float* __restrict__ bskip,
                           float* qbuf, float* skipbuf) {
    int r = blockIdx.x;      // NRQ blocks
    int h = threadIdx.x;     // 64 threads
    float aq = bq[h], as = bskip[h];
    #pragma unroll
    for (int d = 0; d < DRQ; ++d) {
        float x = xr[r * DRQ + d];
        aq += x * Wq[d * HQ + h];
        as += x * Wskip[d * HQ + h];
    }
    qbuf[r * HQ + h] = aq;
    skipbuf[r * HQ + h] = as;
}

// ---------------- K3: order-side WMMA GEMMs: k = xo@Wk+bk, v, p ----------------
__global__ __launch_bounds__(256) void order_gemm(
    const float* __restrict__ xo,
    const _Float16* __restrict__ Wkf, const _Float16* __restrict__ Wvf, const _Float16* __restrict__ Wpf,
    const float* __restrict__ bk, const float* __restrict__ bv, const float* __restrict__ bp,
    float* __restrict__ kout, float* __restrict__ vout, float* __restrict__ pout) {
    const int lane = threadIdx.x & 31;
    const int wave = threadIdx.x >> 5;
    const int tile = blockIdx.x * 8 + wave;
    if (tile >= NOQ / 16) return;            // wave-uniform: EXEC stays all-ones inside
    const int row0 = tile * 16;
    const int m  = lane & 15;                 // row within tile (A)
    const int hi = lane >> 4;                 // K half selector

    // A: 16x64 fp32 tile -> two 16x32 f16 fragments, built from 32B-aligned float4 runs
    v16h a[2];
    #pragma unroll
    for (int kc = 0; kc < 2; ++kc) {
        const v4f* src = (const v4f*)(xo + (size_t)(row0 + m) * DOQ + kc * 32 + hi * 8);
        v4f p0 = src[0], p1 = src[1], p2 = src[4], p3 = src[5];
        v16h t;
        #pragma unroll
        for (int j = 0; j < 4; ++j) {
            t[j]      = (_Float16)p0[j];
            t[4 + j]  = (_Float16)p1[j];
            t[8 + j]  = (_Float16)p2[j];
            t[12 + j] = (_Float16)p3[j];
        }
        a[kc] = t;
    }

    const _Float16* Ws[3]  = { Wkf, Wvf, Wpf };
    const float*    bs[3]  = { bk, bv, bp };
    float*          os[3]  = { kout, vout, pout };

    #pragma unroll
    for (int w = 0; w < 3; ++w) {
        #pragma unroll
        for (int nt = 0; nt < 4; ++nt) {
            v8f acc = {};
            #pragma unroll
            for (int kc = 0; kc < 2; ++kc) {
                // pre-swizzled fragment: one contiguous 32-byte load per lane
                v16h b = *(const v16h*)(Ws[w] + (size_t)(((nt * 2 + kc) * 32 + lane) << 4));
                acc = __builtin_amdgcn_wmma_f32_16x16x32_f16(
                        false, a[kc], false, b, (short)0, acc, false, false);
            }
            #pragma unroll
            for (int r = 0; r < 8; ++r) {
                int row = row0 + r + hi * 8;
                int col = nt * 16 + m;
                os[w][(size_t)row * 64 + col] = acc[r] + bs[w][col];
            }
        }
    }
}

// ---------------- K4: per-edge attention logit -> segment max ----------------
__global__ __launch_bounds__(256) void edge_alpha_max(
    const float* __restrict__ ea, const int* __restrict__ oidx, const int* __restrict__ ridx,
    const float* __restrict__ We, const float* __restrict__ qbuf, const float* __restrict__ kbuf,
    unsigned* rmax) {
    int e = blockIdx.x * blockDim.x + threadIdx.x;
    if (e >= EQ) return;
    int o = oidx[e], r = ridx[e];
    v4f a4 = *(const v4f*)(ea + (size_t)e * 4);
    const v4f* q4 = (const v4f*)(qbuf + (size_t)r * HQ);
    const v4f* k4 = (const v4f*)(kbuf + (size_t)o * HQ);
    const v4f* We0 = (const v4f*)(We);
    const v4f* We1 = (const v4f*)(We + 64);
    const v4f* We2 = (const v4f*)(We + 128);
    const v4f* We3 = (const v4f*)(We + 192);
    float alpha = 0.0f;
    #pragma unroll 4
    for (int h = 0; h < 16; ++h) {
        v4f ev = a4[0] * We0[h] + a4[1] * We1[h] + a4[2] * We2[h] + a4[3] * We3[h];
        v4f kk = k4[h] + ev;
        v4f qq = q4[h];
        alpha += qq[0] * kk[0] + qq[1] * kk[1] + qq[2] * kk[2] + qq[3] * kk[3];
    }
    alpha *= SCALEQ;
    atomicMax(&rmax[r], f2ord(alpha));
}

// ---------------- K5: per-edge exp + numerator/denominator accumulation ----------------
__global__ __launch_bounds__(256) void edge_accum(
    const float* __restrict__ ea, const int* __restrict__ oidx, const int* __restrict__ ridx,
    const float* __restrict__ We, const float* __restrict__ qbuf,
    const float* __restrict__ kbuf, const float* __restrict__ vbuf,
    const unsigned* __restrict__ rmax, float* denom, float* aggnum) {
    int e = blockIdx.x * blockDim.x + threadIdx.x;
    if (e >= EQ) return;
    int o = oidx[e], r = ridx[e];
    v4f a4 = *(const v4f*)(ea + (size_t)e * 4);
    const v4f* q4 = (const v4f*)(qbuf + (size_t)r * HQ);
    const v4f* k4 = (const v4f*)(kbuf + (size_t)o * HQ);
    const v4f* v4 = (const v4f*)(vbuf + (size_t)o * HQ);
    const v4f* We0 = (const v4f*)(We);
    const v4f* We1 = (const v4f*)(We + 64);
    const v4f* We2 = (const v4f*)(We + 128);
    const v4f* We3 = (const v4f*)(We + 192);
    float alpha = 0.0f;
    #pragma unroll 4
    for (int h = 0; h < 16; ++h) {
        v4f ev = a4[0] * We0[h] + a4[1] * We1[h] + a4[2] * We2[h] + a4[3] * We3[h];
        v4f kk = k4[h] + ev;
        v4f qq = q4[h];
        alpha += qq[0] * kk[0] + qq[1] * kk[1] + qq[2] * kk[2] + qq[3] * kk[3];
    }
    alpha *= SCALEQ;
    float m = ord2f(rmax[r]);
    if (m < -1e30f || m > 1e30f) m = 0.0f;       // mirror isfinite() guard
    float ex = __expf(alpha - m);
    atomicAdd(&denom[r], ex);
    float* agg = aggnum + (size_t)r * HQ;
    #pragma unroll 4
    for (int h = 0; h < 16; ++h) {
        v4f ev = a4[0] * We0[h] + a4[1] * We1[h] + a4[2] * We2[h] + a4[3] * We3[h];
        v4f vv = v4[h] + ev;
        atomicAdd(&agg[h * 4 + 0], ex * vv[0]);
        atomicAdd(&agg[h * 4 + 1], ex * vv[1]);
        atomicAdd(&agg[h * 4 + 2], ex * vv[2]);
        atomicAdd(&agg[h * 4 + 3], ex * vv[3]);
    }
}

// ---------------- K6: rider embedding = agg/denom + skip ----------------
__global__ void rider_emb_k(const float* __restrict__ aggnum, const float* __restrict__ denom,
                            const float* __restrict__ skipbuf, float* remb) {
    int r = blockIdx.x, h = threadIdx.x;
    remb[r * HQ + h] = aggnum[r * HQ + h] / (denom[r] + 1e-16f) + skipbuf[r * HQ + h];
}

// ---------------- K7: fused edge MLP via WMMA + dot + clip ----------------
// One wave handles 16 edges: resid_in[16,160](f16) @ W1[160,128](f16) -> relu -> @W2 -> scalar.
// W1 fragments staged into LDS (40KB) once per block via CDNA5 async global->LDS DMA
// (GLOBAL_LOAD_ASYNC_TO_LDS_B128, ASYNCcnt); no VGPR round-trip for the staging data.
__global__ __launch_bounds__(256) void edge_mlp(
    const int* __restrict__ oidx, const int* __restrict__ ridx,
    const float* __restrict__ pbuf, const float* __restrict__ remb,
    const float* __restrict__ omega, const _Float16* __restrict__ W1f,
    const float* __restrict__ b1, const float* __restrict__ W2, const float* __restrict__ b2,
    float* __restrict__ out) {
    __shared__ __align__(32) _Float16 sW1[20480];   // 40 KB
    {
        // LDS byte offset of sW1: shared-aperture flat address, low 32 bits
        unsigned ldsbase = (unsigned)(size_t)(void*)&sW1[0];
        #pragma unroll
        for (int it = 0; it < 10; ++it) {           // 2560 x 16B chunks / 256 threads
            int t = threadIdx.x + it * 256;
            unsigned lds  = ldsbase + (unsigned)t * 16u;
            unsigned goff = (unsigned)t * 16u;
            asm volatile("global_load_async_to_lds_b128 %0, %1, %2"
                         :: "v"(lds), "v"(goff), "s"(W1f)
                         : "memory");
        }
        asm volatile("s_wait_asynccnt 0" ::: "memory");
    }
    __syncthreads();

    const int lane = threadIdx.x & 31;
    const int wave = threadIdx.x >> 5;
    const int tile = blockIdx.x * 8 + wave;
    if (tile >= EQ / 16) return;              // wave-uniform
    const int base = tile * 16;
    const int m  = lane & 15;                 // edge row this lane feeds into A
    const int hi = lane >> 4;

    const int e = base + m;
    const int o = oidx[e];
    const int r = ridx[e];
    const float* oe = pbuf + (size_t)o * HQ;
    const float* re = remb + (size_t)r * HQ;

    // Build A fragments: resid_in = [oe(64) | re(64) | omega(32)], K = 160 = 5 chunks of 32.
    // Each lane's K-run is 4 contiguous 32B float4 spans; keep the oe floats to fuse the
    // dot(oe,re) term (lane hi=0 holds index set {0-7,16-23,32-39,48-55}, hi=1 complement).
    v16h a[5];
    float oef[2][16];
    #pragma unroll
    for (int kc = 0; kc < 2; ++kc) {
        const v4f* src = (const v4f*)(oe + kc * 32 + hi * 8);
        v4f p0 = src[0], p1 = src[1], p2 = src[4], p3 = src[5];
        v16h t;
        #pragma unroll
        for (int j = 0; j < 4; ++j) {
            oef[kc][j]      = p0[j];
            oef[kc][4 + j]  = p1[j];
            oef[kc][8 + j]  = p2[j];
            oef[kc][12 + j] = p3[j];
            t[j]      = (_Float16)p0[j];
            t[4 + j]  = (_Float16)p1[j];
            t[8 + j]  = (_Float16)p2[j];
            t[12 + j] = (_Float16)p3[j];
        }
        a[kc] = t;
    }
    float partial = 0.0f;
    #pragma unroll
    for (int kc = 0; kc < 2; ++kc) {
        const v4f* src = (const v4f*)(re + kc * 32 + hi * 8);
        v4f p0 = src[0], p1 = src[1], p2 = src[4], p3 = src[5];
        v16h t;
        #pragma unroll
        for (int j = 0; j < 4; ++j) {
            partial = fmaf(oef[kc][j],      p0[j], partial);
            partial = fmaf(oef[kc][4 + j],  p1[j], partial);
            partial = fmaf(oef[kc][8 + j],  p2[j], partial);
            partial = fmaf(oef[kc][12 + j], p3[j], partial);
            t[j]      = (_Float16)p0[j];
            t[4 + j]  = (_Float16)p1[j];
            t[8 + j]  = (_Float16)p2[j];
            t[12 + j] = (_Float16)p3[j];
        }
        a[kc + 2] = t;
    }
    {
        const v4f* src = (const v4f*)(omega + hi * 8);
        v4f p0 = src[0], p1 = src[1], p2 = src[4], p3 = src[5];
        v16h t;
        #pragma unroll
        for (int j = 0; j < 4; ++j) {
            t[j]      = (_Float16)p0[j];
            t[4 + j]  = (_Float16)p1[j];
            t[8 + j]  = (_Float16)p2[j];
            t[12 + j] = (_Float16)p3[j];
        }
        a[4] = t;
    }
    // full dot for this lane's edge: combine the two complementary halves
    float dotv = (partial + __shfl_xor(partial, 16, 32)) * SCALEQ;

    // GEMM + fused ReLU + W2 reduction. Lane's columns are c = m + 16*nt.
    float s[8];
    #pragma unroll
    for (int r8 = 0; r8 < 8; ++r8) s[r8] = 0.0f;

    #pragma unroll
    for (int nt = 0; nt < 8; ++nt) {
        v8f acc = {};
        #pragma unroll
        for (int kc = 0; kc < 5; ++kc) {
            v16h b = *(const v16h*)(sW1 + (((nt * 5 + kc) * 32 + lane) << 4));
            acc = __builtin_amdgcn_wmma_f32_16x16x32_f16(
                    false, a[kc], false, b, (short)0, acc, false, false);
        }
        const int c = nt * 16 + m;
        const float bb = b1[c];
        const float ww = W2[c];
        #pragma unroll
        for (int r8 = 0; r8 < 8; ++r8) {
            float hmid = acc[r8] + bb;
            hmid = hmid > 0.0f ? hmid : 0.0f;
            s[r8] = fmaf(hmid, ww, s[r8]);
        }
    }

    // Butterfly reduce across the 16 lanes of each half (rows 0-7 in lanes 0-15, 8-15 in 16-31)
    #pragma unroll
    for (int mask = 1; mask <= 8; mask <<= 1) {
        #pragma unroll
        for (int r8 = 0; r8 < 8; ++r8) s[r8] += __shfl_xor(s[r8], mask, 32);
    }

    const float b2v = b2[0];
    #pragma unroll
    for (int vg = 0; vg < 8; ++vg) {
        float dlo = __shfl(dotv, vg, 32);        // dot of edge row vg
        float dhi = __shfl(dotv, 8 + vg, 32);    // dot of edge row 8+vg
        if (lane == 0) {
            float ov = dlo + s[vg] + b2v;
            out[base + vg] = fminf(10.0f, fmaxf(-10.0f, ov));
        }
        if (lane == 16) {
            float ov = dhi + s[vg] + b2v;
            out[base + 8 + vg] = fminf(10.0f, fmaxf(-10.0f, ov));
        }
    }
}

// ---------------- host launcher ----------------
extern "C" void kernel_launch(void* const* d_in, const int* in_sizes, int n_in,
                              void* d_out, int out_size, void* d_ws, size_t ws_size,
                              hipStream_t stream) {
    const float* x_order  = (const float*)d_in[0];
    const float* x_rider  = (const float*)d_in[1];
    const float* edge_att = (const float*)d_in[2];
    const int*   order_id = (const int*)d_in[3];
    const int*   rider_id = (const int*)d_in[4];
    const float* omega    = (const float*)d_in[5];
    const float* Wk = (const float*)d_in[6];  const float* bk = (const float*)d_in[7];
    const float* Wq = (const float*)d_in[8];  const float* bq = (const float*)d_in[9];
    const float* Wv = (const float*)d_in[10]; const float* bv = (const float*)d_in[11];
    const float* We = (const float*)d_in[12];
    const float* Wskip = (const float*)d_in[13]; const float* bskip = (const float*)d_in[14];
    const float* Wp = (const float*)d_in[15]; const float* bp = (const float*)d_in[16];
    const float* W1 = (const float*)d_in[17]; const float* b1 = (const float*)d_in[18];
    const float* W2 = (const float*)d_in[19]; const float* b2 = (const float*)d_in[20];
    float* out = (float*)d_out;

    // carve workspace
    char* p = (char*)d_ws;
    auto alloc = [&](size_t bytes) -> void* {
        void* ret = (void*)p;
        p += (bytes + 255) & ~(size_t)255;
        return ret;
    };
    float*     kbuf   = (float*)alloc((size_t)NOQ * HQ * 4);
    float*     vbuf   = (float*)alloc((size_t)NOQ * HQ * 4);
    float*     pbuf   = (float*)alloc((size_t)NOQ * HQ * 4);
    float*     qbuf   = (float*)alloc((size_t)NRQ * HQ * 4);
    float*     skipb  = (float*)alloc((size_t)NRQ * HQ * 4);
    float*     remb   = (float*)alloc((size_t)NRQ * HQ * 4);
    float*     aggnum = (float*)alloc((size_t)NRQ * HQ * 4);
    float*     denom  = (float*)alloc((size_t)NRQ * 4);
    unsigned*  rmax   = (unsigned*)alloc((size_t)NRQ * 4);
    _Float16*  Wkf    = (_Float16*)alloc(64 * 64 * 2);
    _Float16*  Wvf    = (_Float16*)alloc(64 * 64 * 2);
    _Float16*  Wpf    = (_Float16*)alloc(64 * 64 * 2);
    _Float16*  W1f    = (_Float16*)alloc(160 * 128 * 2);

    convert_weights<<<32, 256, 0, stream>>>(Wk, Wv, Wp, W1, Wkf, Wvf, Wpf, W1f);
    init_ws<<<(NRQ * HQ + 255) / 256, 256, 0, stream>>>(rmax, denom, aggnum);
    rider_prep<<<NRQ, HQ, 0, stream>>>(x_rider, Wq, bq, Wskip, bskip, qbuf, skipb);
    order_gemm<<<(NOQ / 16 + 7) / 8, 256, 0, stream>>>(x_order, Wkf, Wvf, Wpf, bk, bv, bp,
                                                       kbuf, vbuf, pbuf);
    edge_alpha_max<<<(EQ + 255) / 256, 256, 0, stream>>>(edge_att, order_id, rider_id,
                                                         We, qbuf, kbuf, rmax);
    edge_accum<<<(EQ + 255) / 256, 256, 0, stream>>>(edge_att, order_id, rider_id, We, qbuf,
                                                     kbuf, vbuf, rmax, denom, aggnum);
    rider_emb_k<<<NRQ, HQ, 0, stream>>>(aggnum, denom, skipb, remb);
    edge_mlp<<<(EQ / 16 + 7) / 8, 256, 0, stream>>>(order_id, rider_id, pbuf, remb, omega,
                                                    W1f, b1, W2, b2, out);
    (void)in_sizes; (void)n_in; (void)out_size; (void)ws_size;
}
